// CausalSelfAttention_79877801771239
// MI455X (gfx1250) — matrix operand order repaired and checked by reading, not compiled, not run on hardware
//
#include <hip/hip_runtime.h>
#include <hip/hip_bf16.h>

// ---------------------------------------------------------------------------
// Causal self-attention for B=4, N=2048, C=768, H=8 (d=96) on gfx1250.
// bf16 WMMA (v_wmma_f32_16x16x32_bf16) with f32 accumulation.
//   k_qkv : 4-wave blocks, W tiles staged into LDS by the Tensor Data Mover
//           (tensor_load_to_lds, double-buffered, s_wait_tensorcnt); B-frags
//           preloaded into registers so WMMAs don't serialize on ds waits.
//   k_attn: flash attention, one wave per 16-row q-tile, streaming softmax.
// ---------------------------------------------------------------------------

typedef __attribute__((ext_vector_type(16))) __bf16        v16bf;
typedef __attribute__((ext_vector_type(8)))  float         v8f;
typedef __attribute__((ext_vector_type(4)))  unsigned int  u32x4;
typedef __attribute__((ext_vector_type(8)))  int           i32x8;
typedef __attribute__((ext_vector_type(4)))  int           i32x4;

#define B_  4
#define N_  2048
#define C_  768
#define H_  8
#define D_  96
#define TC_ 2304          // 3*C
#define M_  8192          // B*N

union FragU { v16bf v; u32x4 q[2]; };

__device__ __forceinline__ v8f wmma_bf16(v16bf a, v16bf b, v8f c) {
  return __builtin_amdgcn_wmma_f32_16x16x32_bf16(false, a, false, b, (short)0, c,
                                                 false, false);
}

// A-fragment (16x32 bf16, row-major source with leading dim ld):
// lane L holds row M = L%16; K elements 16*(L>=16) + 0..15 (contiguous).
__device__ __forceinline__ v16bf loadA16(const __bf16* base, int ld, int k0, int lane) {
  const int hi = lane >> 4;
  const __bf16* p = base + (size_t)(lane & 15) * ld + k0 + hi * 16;
  FragU f;
  f.q[0] = *reinterpret_cast<const u32x4*>(p);        // K = k0 + 16*hi + 0..7
  f.q[1] = *reinterpret_cast<const u32x4*>(p + 8);    // K = k0 + 16*hi + 8..15
  return f.v;
}

// B-fragment (32x16 bf16): lane holds column N = L%16; K = 16*(L>=16) + 0..15,
// contiguous at p (caller folds the per-lane column/row offsets).
__device__ __forceinline__ v16bf loadB16(const __bf16* p) {
  FragU f;
  f.q[0] = *reinterpret_cast<const u32x4*>(p);
  f.q[1] = *reinterpret_cast<const u32x4*>(p + 8);
  return f.v;
}

__device__ __forceinline__ float redmax16(float x) {
  x = fmaxf(x, __shfl_xor(x, 1, 32));
  x = fmaxf(x, __shfl_xor(x, 2, 32));
  x = fmaxf(x, __shfl_xor(x, 4, 32));
  x = fmaxf(x, __shfl_xor(x, 8, 32));
  return x;
}

__device__ __forceinline__ float redsum16(float x) {
  x += __shfl_xor(x, 1, 32);
  x += __shfl_xor(x, 2, 32);
  x += __shfl_xor(x, 4, 32);
  x += __shfl_xor(x, 8, 32);
  return x;
}

// ---------------------------------------------------------------------------
// TDM staging of one W tile: 128 rows (output columns) x 64 bytes (32 bf16 of
// K) from Wt (row stride 1536 B) into LDS, rows packed at 64 B.
// D# built per CDNA5 ISA ch.8: 2D tile, data_size=8B, tile_dim0=8, tile_dim1=128.
__device__ __forceinline__ void stage_tile(const __bf16* src, __bf16* dstLds, int tid) {
#if __has_builtin(__builtin_amdgcn_tensor_load_to_lds)
  if (tid < 32) {  // wave 0 issues the DMA once (TDM ignores EXEC; per-wave op)
    unsigned long long ga = (unsigned long long)(uintptr_t)src;
    unsigned lds_off = (unsigned)(uintptr_t)dstLds;  // low 32 bits = LDS offset
    u32x4 g0;
    g0[0] = 1u;                                       // count=1 (valid user D#)
    g0[1] = lds_off;                                  // lds_addr (bytes)
    g0[2] = (unsigned)(ga & 0xffffffffu);             // global_addr[31:0]
    g0[3] = (unsigned)((ga >> 32) & 0x1ffffffu)       // global_addr[56:32]
            | (2u << 30);                             // type=2 ("image")
    const unsigned td0 = 192;                         // tensor_dim0: 1536B row / 8B
    const unsigned td1 = 0x100000;                    // tensor_dim1: large (interior tile)
    const unsigned tl0 = 8;                           // tile_dim0: 8 x 8B = 64B
    const unsigned tl1 = 128;                         // tile_dim1: 128 rows
    const unsigned s0  = 192;                         // dim0 stride, 8B units
    i32x8 g1;
    g1[0] = (int)(3u << 16);                          // data_size=3 (8B); mask=0
    g1[1] = (int)((td0 & 0xffffu) << 16);             // tensor_dim0 lo16
    g1[2] = (int)((td0 >> 16) | ((td1 & 0xffffu) << 16));
    g1[3] = (int)((td1 >> 16) | (tl0 << 16));         // tensor_dim1 hi | tile_dim0
    g1[4] = (int)(tl1);                               // tile_dim1 | tile_dim2=0
    g1[5] = (int)(s0);                                // stride0[31:0]
    g1[6] = 0;                                        // stride0 hi | stride1 lo
    g1[7] = 0;                                        // stride1 hi
    i32x4 gz = {0, 0, 0, 0};
#if defined(__clang_major__) && (__clang_major__ >= 23)
    // clang-23 / therock-10.0 headers: 6-arg form (extra group, then cpol)
    i32x8 gz8 = {0, 0, 0, 0, 0, 0, 0, 0};
    __builtin_amdgcn_tensor_load_to_lds(g0, g1, gz, gz, gz8, 0);
#else
    // ROCm 7.2 clang-22: 5-arg form
    __builtin_amdgcn_tensor_load_to_lds(g0, g1, gz, gz, 0);
#endif
  }
#else
  // Fallback: cooperative copy, 128 threads x 4 x 16B chunks = 8KB.
#pragma unroll
  for (int j = 0; j < 4; ++j) {
    int idx = tid + 128 * j;            // 512 chunks of 16B
    int col = idx >> 2;
    int sub = idx & 3;
    *reinterpret_cast<u32x4*>(dstLds + col * 32 + sub * 8) =
        *reinterpret_cast<const u32x4*>(src + (size_t)col * C_ + sub * 8);
  }
#endif
}

__device__ __forceinline__ void stage_wait(int tid) {
#if __has_builtin(__builtin_amdgcn_tensor_load_to_lds)
  if (tid < 32) __builtin_amdgcn_s_wait_tensorcnt(0);
#endif
  __syncthreads();
}

// ---------------------------------------------------------------------------
// Kernel 0a: x (f32) -> Xb (bf16) row-major [M_, C_]
__global__ void k_cvt_x(const float* __restrict__ x, __bf16* __restrict__ xb, int n) {
  int i = blockIdx.x * blockDim.x + threadIdx.x;
  int stride = gridDim.x * blockDim.x;
  for (; i < n; i += stride) xb[i] = (__bf16)x[i];
}

// Kernel 0b: W [C_, TC_] f32 -> Wt [TC_, C_] bf16 (transposed)
__global__ void k_tr_w(const float* __restrict__ W, __bf16* __restrict__ wt) {
  int i = blockIdx.x * blockDim.x + threadIdx.x;
  if (i >= TC_ * C_) return;
  int c = i / C_;
  int k = i - c * C_;
  wt[i] = (__bf16)W[(size_t)k * TC_ + c];
}

// ---------------------------------------------------------------------------
// Kernel 1: QKV projection. Block = 4 waves computing a 64x128 tile of
// qkv = Xb @ W + b. W tile (128x32 k-slice, 8KB) is TDM-staged into LDS and
// double-buffered; each wave owns 16 rows. All 8 B-fragments are preloaded
// into registers before the WMMA chain so the ds loads pipeline under the
// WMMAs instead of each WMMA stalling on s_wait_dscnt 0.
__global__ void __launch_bounds__(128)
k_qkv(const __bf16* __restrict__ xb, const __bf16* __restrict__ wt,
      const float* __restrict__ bias,
      __bf16* __restrict__ qb, __bf16* __restrict__ kb, __bf16* __restrict__ vt) {
  __shared__ __align__(16) __bf16 ldsW[2][128 * 32];   // 2 x 8KB

  const int tid  = threadIdx.x;
  const int lane = tid & 31;
  const int w    = tid >> 5;
  const int hi   = lane >> 4;
  const int c0   = blockIdx.x * 128;
  const int r0   = blockIdx.y * 64 + w * 16;

  const v8f vzero = {0.f, 0.f, 0.f, 0.f, 0.f, 0.f, 0.f, 0.f};
  v8f acc[8];
#pragma unroll
  for (int t = 0; t < 8; ++t) acc[t] = vzero;

  const __bf16* arow = xb + (size_t)r0 * C_;
  const __bf16* wcol = wt + (size_t)c0 * C_;

  stage_tile(wcol, &ldsW[0][0], tid);
  stage_wait(tid);

  int ib = 0;
  for (int kc = 0; kc < C_; kc += 32, ib ^= 1) {
    if (kc + 32 < C_) stage_tile(wcol + (kc + 32), &ldsW[ib ^ 1][0], tid);

    v16bf a = loadA16(arow, C_, kc, lane);
    const __bf16* bbase = &ldsW[ib][0] + (size_t)(lane & 15) * 32 + hi * 16;

    v16bf bw[8];                 // preload all B-fragments (16 ds_load_b128)
#pragma unroll
    for (int t = 0; t < 8; ++t) bw[t] = loadB16(bbase + t * 16 * 32);
#pragma unroll
    for (int t = 0; t < 8; ++t) acc[t] = wmma_bf16(a, bw[t], acc[t]);

    if (kc + 32 < C_) stage_wait(tid);  // next buffer landed; prev reads done
  }

  const float qscale = 0.10206207261596575f;  // 1/sqrt(96)
#pragma unroll
  for (int t = 0; t < 8; ++t) {
    const int col = c0 + t * 16 + (lane & 15);
#pragma unroll
    for (int r = 0; r < 8; ++r) {
      const int g  = r0 + r + hi * 8;      // global row in [0, M_)
      const int bi = g >> 11;              // batch
      const int n  = g & (N_ - 1);         // sequence pos
      float v = acc[t][r] + bias[col];
      if (col < C_) {
        int h = col / D_, dd = col - h * D_;
        qb[((size_t)(bi * H_ + h) * N_ + n) * D_ + dd] = (__bf16)(v * qscale);
      } else if (col < 2 * C_) {
        int cc = col - C_;
        int h = cc / D_, dd = cc - h * D_;
        kb[((size_t)(bi * H_ + h) * N_ + n) * D_ + dd] = (__bf16)v;
      } else {
        int cc = col - 2 * C_;
        int h = cc / D_, dd = cc - h * D_;
        vt[((size_t)(bi * H_ + h) * D_ + dd) * N_ + n] = (__bf16)v;
      }
    }
  }
}

// ---------------------------------------------------------------------------
// Kernel 2: flash attention. One wave owns one 16-row q-tile of one (b,h).
// 8 independent waves per block (no cross-wave barriers: causal trip counts
// diverge). Per 32-key kv tile: 6 WMMAs for S, register softmax with 16-lane
// row reductions, P -> LDS (C-layout -> A-layout), 6 WMMAs for P@V.
__global__ void __launch_bounds__(256)
k_attn(const __bf16* __restrict__ qb, const __bf16* __restrict__ kb,
       const __bf16* __restrict__ vt, float* __restrict__ out) {
  __shared__ __align__(16) __bf16 ldsP[8][16 * 32];

  const int lane = threadIdx.x & 31;
  const int w    = threadIdx.x >> 5;
  const int hi   = lane >> 4;
  const int id   = blockIdx.x * 8 + w;      // [0, B*H*N/16)
  const int qt   = id & 127;                // N/16 = 128
  const int bh   = id >> 7;
  const int q0   = qt * 16;

  const __bf16* Q = qb + (size_t)bh * N_ * D_;
  const __bf16* K = kb + (size_t)bh * N_ * D_;
  const __bf16* V = vt + (size_t)bh * D_ * N_;

  // Q tile (already scaled by 1/sqrt(d)): three 16x32 A-fragments.
  v16bf qA0 = loadA16(Q + (size_t)q0 * D_, D_, 0,  lane);
  v16bf qA1 = loadA16(Q + (size_t)q0 * D_, D_, 32, lane);
  v16bf qA2 = loadA16(Q + (size_t)q0 * D_, D_, 64, lane);

  const v8f vzero = {0.f, 0.f, 0.f, 0.f, 0.f, 0.f, 0.f, 0.f};
  v8f acc[6];
#pragma unroll
  for (int t = 0; t < 6; ++t) acc[t] = vzero;
  float m[8], l[8];
#pragma unroll
  for (int r = 0; r < 8; ++r) { m[r] = -1e30f; l[r] = 0.f; }

  __bf16* lp = &ldsP[w][0];
  const int ntile = (q0 + 47) >> 5;  // kv tiles of 32 covering keys <= q0+15

  for (int kt = 0; kt < ntile; ++kt) {
    const int key0 = kt * 32;

    // --- S = Q @ K^T for keys [key0, key0+32) -----------------------------
    v8f s0 = vzero, s1 = vzero;
    const __bf16* kp0 = K + (size_t)(key0 + (lane & 15)) * D_ + hi * 16;
    s0 = wmma_bf16(qA0, loadB16(kp0 + 0),  s0);
    s0 = wmma_bf16(qA1, loadB16(kp0 + 32), s0);
    s0 = wmma_bf16(qA2, loadB16(kp0 + 64), s0);
    const __bf16* kp1 = kp0 + (size_t)16 * D_;
    s1 = wmma_bf16(qA0, loadB16(kp1 + 0),  s1);
    s1 = wmma_bf16(qA1, loadB16(kp1 + 32), s1);
    s1 = wmma_bf16(qA2, loadB16(kp1 + 64), s1);

    // --- causal mask (boundary tiles only) --------------------------------
    if (key0 + 15 > q0) {
      const int keyc = key0 + (lane & 15);
#pragma unroll
      for (int r = 0; r < 8; ++r) {
        if (keyc > q0 + r + hi * 8) s0[r] = -1e30f;
      }
    }
    if (key0 + 31 > q0) {
      const int keyc = key0 + 16 + (lane & 15);
#pragma unroll
      for (int r = 0; r < 8; ++r) {
        if (keyc > q0 + r + hi * 8) s1[r] = -1e30f;
      }
    }

    // --- online softmax update --------------------------------------------
#pragma unroll
    for (int r = 0; r < 8; ++r) {
      float mt = redmax16(fmaxf(s0[r], s1[r]));
      float mn = fmaxf(m[r], mt);
      float fs = __expf(m[r] - mn);
      float p0 = __expf(s0[r] - mn);
      float p1 = __expf(s1[r] - mn);
      s0[r] = p0;
      s1[r] = p1;
      l[r] = l[r] * fs + redsum16(p0 + p1);
      m[r] = mn;
#pragma unroll
      for (int t = 0; t < 6; ++t) acc[t][r] *= fs;
    }

    // --- P: C-layout -> A-layout via per-wave LDS tile (row-major 16x32) --
#pragma unroll
    for (int r = 0; r < 8; ++r) {
      const int row = r + hi * 8;
      lp[row * 32 + (lane & 15)]      = (__bf16)s0[r];
      lp[row * 32 + 16 + (lane & 15)] = (__bf16)s1[r];
    }
    __threadfence_block();  // order LDS stores before loads (in-wave)
    v16bf pA = loadA16(lp, 32, 0, lane);

    // --- O += P @ V --------------------------------------------------------
    const __bf16* vp = V + (size_t)(lane & 15) * N_ + key0 + hi * 16;
#pragma unroll
    for (int t = 0; t < 6; ++t) {
      acc[t] = wmma_bf16(pA, loadB16(vp + (size_t)t * 16 * N_), acc[t]);
    }
  }

  // --- epilogue: normalize and write [B, N, C] with merged heads ----------
  const int b = bh >> 3;
  const int h = bh & 7;
#pragma unroll
  for (int r = 0; r < 8; ++r) {
    const float inv = 1.0f / l[r];
    const int row = q0 + r + hi * 8;
    float* orow = out + ((size_t)(b * N_ + row)) * C_ + h * D_ + (lane & 15);
#pragma unroll
    for (int t = 0; t < 6; ++t) orow[t * 16] = acc[t][r] * inv;
  }
}

// ---------------------------------------------------------------------------
extern "C" void kernel_launch(void* const* d_in, const int* in_sizes, int n_in,
                              void* d_out, int out_size, void* d_ws, size_t ws_size,
                              hipStream_t stream) {
  const float* x    = (const float*)d_in[0];   // [B, N, C]
  const float* W    = (const float*)d_in[1];   // [C, 3C]
  const float* bias = (const float*)d_in[2];   // [3C]
  float* out        = (float*)d_out;           // [B, N, C]

  char* ws = (char*)d_ws;
  // Fixed carve (16B-aligned): Xb | Wt | Q | K | Vt  (~54 MB total)
  __bf16* xb = (__bf16*)(ws + 0);                     // M_*C_   bf16 = 12.58 MB
  __bf16* wt = (__bf16*)(ws + 12582912);              // TC_*C_  bf16 =  3.54 MB
  __bf16* qb = (__bf16*)(ws + 16121856);              // B*H*N*D bf16 = 12.58 MB
  __bf16* kb = (__bf16*)(ws + 28704768);              // B*H*N*D bf16 = 12.58 MB
  __bf16* vt = (__bf16*)(ws + 41287680);              // B*H*D*N bf16 = 12.58 MB

  k_cvt_x<<<dim3((M_ * C_) / 256), 256, 0, stream>>>(x, xb, M_ * C_);
  k_tr_w<<<dim3((TC_ * C_ + 255) / 256), 256, 0, stream>>>(W, wt);
  k_qkv<<<dim3(TC_ / 128, M_ / 64), 128, 0, stream>>>(xb, wt, bias, qb, kb, vt);
  k_attn<<<dim3((B_ * H_ * (N_ / 16)) / 8), 256, 0, stream>>>(qb, kb, vt, out);
}
